// ComposableMoE_90735479095889
// MI455X (gfx1250) — matrix operand order repaired and checked by reading, not compiled, not run on hardware
//
#include <hip/hip_runtime.h>
#include <hip/hip_bf16.h>

typedef __attribute__((ext_vector_type(16))) _Float16 v16h;
typedef __attribute__((ext_vector_type(8)))  float    v8f;
typedef __attribute__((ext_vector_type(4)))  float    v4f;

#define NTOK 16384
#define DDIM 512
#define NEXP 16
#define EMBD 128
#define HDIM 1024
#define H2D  512

constexpr int TN = 64, TK = 32;
constexpr int BSTR = 36;                     // LDS col stride (halves), padded vs bank conflicts
constexpr int RTM = 64;                      // router block rows (4 waves x 16)
constexpr int EM  = 128;                     // expert gemm block rows (4 waves x 32)
constexpr int CAP  = NTOK * 2 + NEXP * EM;   // 34816 compacted rows incl. per-expert pad
constexpr int CAPT = CAP / EM;               // 272 row tiles

// ---------------- workspace layout (bytes) ----------------
constexpr size_t WS_CNT   = 0;                                   // 16 ints
constexpr size_t WS_FILL  = 64;                                  // 16 ints
constexpr size_t WS_PST   = 128;                                 // 17 ints
constexpr size_t WS_TOPI  = 256;                                 // N*2 ints
constexpr size_t WS_GATES = WS_TOPI  + (size_t)NTOK * 2 * 4;     // N*2 floats
constexpr size_t WS_TLIST = WS_GATES + (size_t)NTOK * 2 * 4;     // CAP ints
constexpr size_t WS_GLIST = WS_TLIST + (size_t)CAP * 4;          // CAP floats
constexpr size_t WS_H1    = (WS_GLIST + (size_t)CAP * 4 + 255) & ~(size_t)255; // CAP*H f16
constexpr size_t WS_H2    = WS_H1 + (size_t)CAP * HDIM * 2;      // CAP*H2 f16

// ---------------- WMMA fragment helpers ----------------
// A (16x32 f16): lane<16 -> row=lane,   K 0-7 in a[0..7],  K 16-23 in a[8..15]
//                lane>=16 -> row=lane-16, K 8-15 / 24-31   (kb = 8 halves)
__device__ inline v16h load_afrag_f32(const float* __restrict__ rowp, int k0, int kb) {
    v16h a;
#pragma unroll
    for (int i = 0; i < 8; ++i) {
        a[i]     = (_Float16)rowp[k0 + kb + i];
        a[i + 8] = (_Float16)rowp[k0 + kb + 16 + i];
    }
    return a;
}
__device__ inline v16h load_afrag_f16(const _Float16* __restrict__ rowp, int k0, int kb) {
    v16h a;
#pragma unroll
    for (int i = 0; i < 8; ++i) {
        a[i]     = rowp[k0 + kb + i];
        a[i + 8] = rowp[k0 + kb + 16 + i];
    }
    return a;
}

// Stage B tile (TK=32 rows of K, TN=64 cols) f32 -> f16 into LDS, column-major [col][k]
__device__ inline void stage_b(const float* __restrict__ W, int nout, int k0, int n0,
                               _Float16* Bs, int tid) {
#pragma unroll
    for (int i = 0; i < 4; ++i) {
        int f  = tid + i * 128;       // 0..511, each covers a float4
        int k  = f >> 4;              // 0..31
        int c4 = (f & 15) * 4;        // 0,4,...,60
        const v4f* p = reinterpret_cast<const v4f*>(W + (size_t)(k0 + k) * nout + n0 + c4);
        v4f w = *p;
#pragma unroll
        for (int u = 0; u < 4; ++u)
            Bs[(c4 + u) * BSTR + k] = (_Float16)w[u];
    }
}

// B (32x16 f16): lane<16 -> col=lane, K 0-7 / 16-23; lane>=16 -> col=lane-16, K 8-15 / 24-31
__device__ inline v16h load_bfrag(const _Float16* Bs, int lane, int g) {
    int col = g * 16 + (lane & 15);
    int kb  = (lane >= 16) ? 8 : 0;
    const _Float16* p = Bs + col * BSTR + kb;
    v16h b;
#pragma unroll
    for (int i = 0; i < 8; ++i) { b[i] = p[i]; b[i + 8] = p[16 + i]; }
    return b;
}

#define WMMA_F16(C, A, B) \
    __builtin_amdgcn_wmma_f32_16x16x32_f16(false, (A), false, (B), (short)0, (C), false, false)

// ---------------- kernels ----------------
__global__ void init_kernel(float* __restrict__ combined, int* __restrict__ cnt,
                            int* __restrict__ fill, int* __restrict__ tok_list) {
    int idx = blockIdx.x * blockDim.x + threadIdx.x;
    int stride = gridDim.x * blockDim.x;
    for (int i = idx; i < NTOK * DDIM; i += stride) combined[i] = 0.f;
    for (int i = idx; i < CAP; i += stride) tok_list[i] = -1;
    if (idx < NEXP) { cnt[idx] = 0; fill[idx] = 0; }
}

__global__ void __launch_bounds__(128) router_gemm(
    const float* __restrict__ X, const float* __restrict__ Wr,
    const float* __restrict__ br, float* __restrict__ q) {
    __shared__ _Float16 Bs[2][TN * BSTR];
    const int r0 = blockIdx.x * RTM, n0 = blockIdx.y * TN;
    const int tid = threadIdx.x, lane = tid & 31, wave = tid >> 5;
    const float* arow = X + (size_t)(r0 + wave * 16 + (lane & 15)) * DDIM;
    const int kb = (lane >= 16) ? 8 : 0;
    v8f c[4] = {};
    constexpr int NS = DDIM / TK;
    stage_b(Wr, EMBD, 0, n0, Bs[0], tid);
    __syncthreads();
    int buf = 0;
    for (int s = 0; s < NS; ++s) {
        int k0 = s * TK;
        if (s + 1 < NS) stage_b(Wr, EMBD, k0 + TK, n0, Bs[buf ^ 1], tid);
        v16h a = load_afrag_f32(arow, k0, kb);
#pragma unroll
        for (int g = 0; g < 4; ++g)
            c[g] = WMMA_F16(c[g], a, load_bfrag(Bs[buf], lane, g));
        __syncthreads();
        buf ^= 1;
    }
    const int mo = (lane >= 16) ? 8 : 0;
#pragma unroll
    for (int g = 0; g < 4; ++g) {
        int col = n0 + g * 16 + (lane & 15);
        float bv = br[col];
#pragma unroll
        for (int j = 0; j < 8; ++j) {
            int row = r0 + wave * 16 + j + mo;
            q[(size_t)row * EMBD + col] = c[g][j] + bv;
        }
    }
}

__global__ void __launch_bounds__(32) scores_kernel(
    const float* __restrict__ q, const float* __restrict__ emb,
    float* __restrict__ gp, int* __restrict__ topi, float* __restrict__ gates) {
    const int n = blockIdx.x, lane = threadIdx.x, e = lane & 15;
    const float* qr = q + (size_t)n * EMBD;
    const float* er = emb + (size_t)e * EMBD;
    float acc = 0.f;
    for (int j = 0; j < EMBD; ++j) { float d = qr[j] - er[j]; acc = fmaf(d, d, acc); }
    float score = -acc;
    float m = score;
    for (int off = 1; off < 16; off <<= 1) m = fmaxf(m, __shfl_xor(m, off, 32));
    float p = __expf(score - m);
    float s = p;
    for (int off = 1; off < 16; off <<= 1) s += __shfl_xor(s, off, 32);
    if (lane < 16) gp[(size_t)n * NEXP + e] = p / s;
    float m1 = m;
    int i1 = (score == m1) ? e : 1000;
    for (int off = 1; off < 16; off <<= 1) { int t = __shfl_xor(i1, off, 32); i1 = t < i1 ? t : i1; }
    float s2 = (e == i1) ? -3.0e38f : score;
    float m2 = s2;
    for (int off = 1; off < 16; off <<= 1) m2 = fmaxf(m2, __shfl_xor(m2, off, 32));
    int i2 = (s2 == m2) ? e : 1000;
    for (int off = 1; off < 16; off <<= 1) { int t = __shfl_xor(i2, off, 32); i2 = t < i2 ? t : i2; }
    if (lane == 0) {
        topi[2 * n] = i1; topi[2 * n + 1] = i2;
        float g1 = 1.f / (1.f + __expf(m2 - m1));
        gates[2 * n] = g1; gates[2 * n + 1] = 1.f - g1;
    }
}

__global__ void count_kernel(const int* __restrict__ topi, int* __restrict__ cnt) {
    int n = blockIdx.x * blockDim.x + threadIdx.x;
    if (n < NTOK) { atomicAdd(&cnt[topi[2 * n]], 1); atomicAdd(&cnt[topi[2 * n + 1]], 1); }
}

__global__ void prefix_kernel(const int* __restrict__ cnt, int* __restrict__ pstart) {
    if (threadIdx.x == 0 && blockIdx.x == 0) {
        int s = 0;
        for (int e = 0; e < NEXP; ++e) { pstart[e] = s; s += (cnt[e] + EM - 1) / EM * EM; }
        pstart[NEXP] = s;
    }
}

__global__ void fill_kernel(const int* __restrict__ topi, const float* __restrict__ gates,
                            const int* __restrict__ pstart, int* __restrict__ fillc,
                            int* __restrict__ tok_list, float* __restrict__ gate_list) {
    int n = blockIdx.x * blockDim.x + threadIdx.x;
    if (n >= NTOK) return;
    for (int k = 0; k < 2; ++k) {
        int e = topi[2 * n + k];
        int pos = atomicAdd(&fillc[e], 1);
        int row = pstart[e] + pos;
        tok_list[row] = n;
        gate_list[row] = gates[2 * n + k];
    }
}

// Routed expert GEMM: L=1 X->h1(relu), L=2 h1->h2(relu), L=3 h2->combined (gated atomic)
// Block tile 128x64 (4 waves, 32 rows each -> 2 A frags, 8 WMMA per K-step per wave)
template <int L>
__global__ void __launch_bounds__(128) expert_gemm(
    const float* __restrict__ X, const _Float16* __restrict__ Hin,
    const float* __restrict__ Wall, const float* __restrict__ Ball,
    _Float16* __restrict__ Hout, float* __restrict__ combined,
    const int* __restrict__ pstart, const int* __restrict__ tok_list,
    const float* __restrict__ gate_list) {
    constexpr int KD = (L == 1) ? DDIM : (L == 2 ? HDIM : H2D);
    constexpr int NO = (L == 1) ? HDIM : (L == 2 ? H2D : DDIM);
    constexpr int NS = KD / TK;

    __shared__ _Float16 Bs[2][TN * BSTR];
    const int r0 = blockIdx.x * EM;
    if (r0 >= pstart[NEXP]) return;           // beyond last routed tile
    int e = 0;
    while (r0 >= pstart[e + 1]) ++e;          // expert owning this row tile

    const int n0 = blockIdx.y * TN;
    const int tid = threadIdx.x, lane = tid & 31, wave = tid >> 5;
    const int mrow0 = r0 + wave * 32 + (lane & 15);       // A frag 0 row
    const int mrow1 = mrow0 + 16;                          // A frag 1 row

    const float* W    = Wall + (size_t)e * KD * NO;
    const float* bias = Ball + (size_t)e * NO;

    const float*    a32[2] = {nullptr, nullptr};
    const _Float16* a16[2] = {nullptr, nullptr};
    if constexpr (L == 1) {
        int t0 = tok_list[mrow0], t1 = tok_list[mrow1];
        if (t0 >= 0) a32[0] = X + (size_t)t0 * DDIM;
        if (t1 >= 0) a32[1] = X + (size_t)t1 * DDIM;
    } else {
        a16[0] = Hin + (size_t)mrow0 * KD;
        a16[1] = Hin + (size_t)mrow1 * KD;
    }
    const int kb = (lane >= 16) ? 8 : 0;

    v8f c[2][4] = {};
    stage_b(W, NO, 0, n0, Bs[0], tid);
    __syncthreads();
    int buf = 0;
    for (int s = 0; s < NS; ++s) {
        int k0 = s * TK;
        if (s + 1 < NS) stage_b(W, NO, k0 + TK, n0, Bs[buf ^ 1], tid);
        if (s + 2 < NS)   // warm L2/L0 for the K-tile after next (global_prefetch_b8)
            __builtin_prefetch(W + (size_t)(k0 + 2 * TK + (tid >> 2)) * NO + n0 + (tid & 3) * 16, 0, 1);
        v16h a0{}, a1{};
        if constexpr (L == 1) {
            if (a32[0]) a0 = load_afrag_f32(a32[0], k0, kb);
            if (a32[1]) a1 = load_afrag_f32(a32[1], k0, kb);
        } else {
            a0 = load_afrag_f16(a16[0], k0, kb);
            a1 = load_afrag_f16(a16[1], k0, kb);
        }
#pragma unroll
        for (int g = 0; g < 4; ++g) {
            v16h b = load_bfrag(Bs[buf], lane, g);
            c[0][g] = WMMA_F16(c[0][g], a0, b);
            c[1][g] = WMMA_F16(c[1][g], a1, b);
        }
        __syncthreads();
        buf ^= 1;
    }

    const int mo = (lane >= 16) ? 8 : 0;
#pragma unroll
    for (int f = 0; f < 2; ++f) {
#pragma unroll
        for (int g = 0; g < 4; ++g) {
            int col = n0 + g * 16 + (lane & 15);
            float bv = bias[col];
#pragma unroll
            for (int j = 0; j < 8; ++j) {
                int row = r0 + wave * 32 + f * 16 + j + mo;
                float v = c[f][g][j] + bv;
                if constexpr (L == 3) {
                    int tok = tok_list[row];
                    if (tok >= 0)
                        atomicAdd(combined + (size_t)tok * DDIM + col, v * gate_list[row]);
                } else {
                    v = v > 0.f ? v : 0.f;
                    Hout[(size_t)row * NO + col] = (_Float16)v;
                }
            }
        }
    }
}

extern "C" void kernel_launch(void* const* d_in, const int* in_sizes, int n_in,
                              void* d_out, int out_size, void* d_ws, size_t ws_size,
                              hipStream_t stream) {
    const float* X    = (const float*)d_in[0];
    const float* Wr   = (const float*)d_in[1];
    const float* br   = (const float*)d_in[2];
    const float* emb  = (const float*)d_in[3];
    const float* W1   = (const float*)d_in[4];
    const float* b1   = (const float*)d_in[5];
    const float* W2   = (const float*)d_in[6];
    const float* b2   = (const float*)d_in[7];
    const float* W3   = (const float*)d_in[8];
    const float* b3   = (const float*)d_in[9];

    float* combined = (float*)d_out;
    float* qout     = combined + (size_t)NTOK * DDIM;
    float* gpout    = qout + (size_t)NTOK * EMBD;

    char* ws = (char*)d_ws;
    int*      cnt       = (int*)(ws + WS_CNT);
    int*      fillc     = (int*)(ws + WS_FILL);
    int*      pstart    = (int*)(ws + WS_PST);
    int*      topi      = (int*)(ws + WS_TOPI);
    float*    gates     = (float*)(ws + WS_GATES);
    int*      tok_list  = (int*)(ws + WS_TLIST);
    float*    gate_list = (float*)(ws + WS_GLIST);
    _Float16* h1        = (_Float16*)(ws + WS_H1);
    _Float16* h2        = (_Float16*)(ws + WS_H2);

    init_kernel<<<2048, 256, 0, stream>>>(combined, cnt, fillc, tok_list);
    router_gemm<<<dim3(NTOK / RTM, EMBD / TN), 128, 0, stream>>>(X, Wr, br, qout);
    scores_kernel<<<NTOK, 32, 0, stream>>>(qout, emb, gpout, topi, gates);
    count_kernel<<<NTOK / 256, 256, 0, stream>>>(topi, cnt);
    prefix_kernel<<<1, 1, 0, stream>>>(cnt, pstart);
    fill_kernel<<<NTOK / 256, 256, 0, stream>>>(topi, gates, pstart, fillc, tok_list, gate_list);
    expert_gemm<1><<<dim3(CAPT, HDIM / TN), 128, 0, stream>>>(
        X, nullptr, W1, b1, h1, nullptr, pstart, tok_list, gate_list);
    expert_gemm<2><<<dim3(CAPT, H2D / TN), 128, 0, stream>>>(
        nullptr, h1, W2, b2, h2, nullptr, pstart, tok_list, gate_list);
    expert_gemm<3><<<dim3(CAPT, DDIM / TN), 128, 0, stream>>>(
        nullptr, h2, W3, b3, nullptr, combined, pstart, tok_list, gate_list);
    (void)in_sizes; (void)n_in; (void)out_size; (void)ws_size;
}
